// Generator_28836410426135
// MI455X (gfx1250) — matrix operand order repaired and checked by reading, not compiled, not run on hardware
//
#include <hip/hip_runtime.h>
#include <stdint.h>

#define LEAK  0.2f
#define SQRT2 1.4142135623730951f

typedef __attribute__((ext_vector_type(16))) _Float16 v16h;
typedef __attribute__((ext_vector_type(8)))  _Float16 v8h;
typedef __attribute__((ext_vector_type(8)))  float    v8f;
typedef __attribute__((ext_vector_type(4)))  unsigned int u32x4;
typedef __attribute__((ext_vector_type(8)))  int      i32x8;
typedef __attribute__((ext_vector_type(4)))  int      i32x4;

// ---------------- workspace layout (bytes) ----------------
#define OFF_S1  ((size_t)0)                       // 8*512 f32
#define OFF_S2  ((size_t)16384)                   // 8*256 f32
#define OFF_D1  ((size_t)24576)                   // 8*256 f32
#define OFF_D2  ((size_t)32768)                   // 8*256 f32
#define OFF_XS  ((size_t)65536)                   // 8*512*64*64 f16   = 33554432 B
#define OFF_W1  (OFF_XS + (size_t)33554432)       // 4*256*4608 f16    =  9437184 B
#define OFF_W2  (OFF_W1 + (size_t)9437184)        // 256*2304 f16      =  1179648 B
#define OFF_Y1  (OFF_W2 + (size_t)1179648)        // 8*256*128*128 f16 = 67108864 B

// ---------------- small prep kernels ----------------

// s1[b,i] = w1[b]·aff1_w[i] + aff1_b[i] + 1 ;  s2[b,c] likewise
__global__ void k_affine(const float* __restrict__ w1, const float* __restrict__ w2,
                         const float* __restrict__ a1w, const float* __restrict__ a1b,
                         const float* __restrict__ a2w, const float* __restrict__ a2b,
                         float* __restrict__ s1, float* __restrict__ s2) {
  int idx = blockIdx.x * 256 + threadIdx.x;
  if (idx < 8 * 512) {
    int b = idx >> 9, i = idx & 511;
    float acc = a1b[i] + 1.0f;
    for (int l = 0; l < 512; ++l) acc += w1[b * 512 + l] * a1w[i * 512 + l];
    s1[idx] = acc;
  } else {
    int j = idx - 8 * 512;
    if (j < 8 * 256) {
      int b = j >> 8, c = j & 255;
      float acc = a2b[c] + 1.0f;
      for (int l = 0; l < 512; ++l) acc += w2[b * 512 + l] * a2w[c * 512 + l];
      s2[j] = acc;
    }
  }
}

// d[b,o] = rsqrt( sum_i s[b,i]^2 * sum_k w[o,i,k]^2 + 1e-8 )
__global__ void k_demod(const float* __restrict__ s1, const float* __restrict__ s2,
                        const float* __restrict__ wt1, const float* __restrict__ wt2,
                        float* __restrict__ d1, float* __restrict__ d2) {
  int idx = blockIdx.x * 256 + threadIdx.x;   // 0..4095
  int half = idx >> 11;
  int j = idx & 2047;
  int b = j >> 8, o = j & 255;
  float acc = 1e-8f;
  if (half == 0) {
    for (int i = 0; i < 512; ++i) {
      const float* p = wt1 + ((size_t)o * 512 + i) * 9;
      float ws = 0.f;
      #pragma unroll
      for (int t = 0; t < 9; ++t) ws += p[t] * p[t];
      float s = s1[b * 512 + i];
      acc += s * s * ws;
    }
    d1[j] = rsqrtf(acc);
  } else {
    for (int i = 0; i < 256; ++i) {
      const float* p = wt2 + ((size_t)o * 256 + i) * 9;
      float ws = 0.f;
      #pragma unroll
      for (int t = 0; t < 9; ++t) ws += p[t] * p[t];
      float s = s2[b * 256 + i];
      acc += s * s * ws;
    }
    d2[j] = rsqrtf(acc);
  }
}

// xs = f16( x * s1[b,i] )
__global__ void k_scale_x(const float* __restrict__ x, const float* __restrict__ s1,
                          _Float16* __restrict__ xs) {
  size_t e = (size_t)blockIdx.x * 256 + threadIdx.x;  // 16,777,216 total
  int bi = (int)(e >> 12);                            // b*512 + i
  xs[e] = (_Float16)(x[e] * s1[bi]);
}

// Effective upsample+blur kernel per output parity (ph,pw):
// W1eff[p][o][tap*512+i] = sum_{kh,kw} w1[o,i,kh,kw] * bk[kh+1-eh] * bk[kw+1-ew]
// eh = (ph==0 ? 2-2*dh : 3-2*dh), taps gather input rows {t-1,t,t+1} (pad 1).
__global__ void k_build_w1(const float* __restrict__ wt1, _Float16* __restrict__ W1) {
  int idx = blockIdx.x * 256 + threadIdx.x;  // 4*9*512*256
  int o = idx & 255;
  int r = idx >> 8;
  int i = r & 511; r >>= 9;
  int tap = r % 9; int p = r / 9;
  if (p >= 4) return;
  int dh = tap / 3, dw = tap % 3;
  int ph = p >> 1, pw = p & 1;
  int eh = (ph == 0) ? (2 - 2 * dh) : (3 - 2 * dh);
  int ew = (pw == 0) ? (2 - 2 * dw) : (3 - 2 * dw);
  const float bk[4] = {0.25f, 0.75f, 0.75f, 0.25f};  // [1,3,3,1]/4 (incl. up^2 gain)
  const float* wp = wt1 + ((size_t)o * 512 + i) * 9;
  float acc = 0.f;
  #pragma unroll
  for (int kh = 0; kh < 3; ++kh) {
    int ih = kh + 1 - eh;
    if (ih < 0 || ih > 3) continue;
    #pragma unroll
    for (int kw = 0; kw < 3; ++kw) {
      int iw = kw + 1 - ew;
      if (iw < 0 || iw > 3) continue;
      acc += wp[kh * 3 + kw] * bk[ih] * bk[iw];
    }
  }
  W1[(((size_t)p * 256 + o) * 4608) + tap * 512 + i] = (_Float16)acc;
}

// W2mat[o][tap*256+i] = f16(weight2[o,i,tap])
__global__ void k_build_w2(const float* __restrict__ wt2, _Float16* __restrict__ W2) {
  int idx = blockIdx.x * 256 + threadIdx.x;  // 256*2304
  int o = idx / 2304;
  int rem = idx - o * 2304;
  int tap = rem >> 8, i = rem & 255;
  W2[idx] = (_Float16)wt2[((size_t)o * 256 + i) * 9 + tap];
}

// ---------------- WMMA implicit-GEMM conv (double-buffered, TDM-pipelined) ----
// mode 0: conv1 (fused upsample+blur, per-parity 3x3 over 64x64, Cin=512, Ktot=4608)
//         epilogue: d1*acc + ns1*noise1 + b1, lrelu*sqrt2, *s2  -> y1 f16 (strided 2)
// mode 1: conv2 (3x3 pad1 over 128x128, Cin=256, Ktot=2304)
//         epilogue: d2*acc + ns2*noise2 + b2, lrelu*sqrt2       -> d_out f32
//
// LDS: two buffers of 12288 B each. Buffer q: A tile [64][32] f16 at byte
// q*12288, B tile [128][32] f16 ([n][k]) at byte q*12288 + 4096.
__global__ void __launch_bounds__(256)
conv_wmma(int mode,
          const _Float16* __restrict__ inp,
          const _Float16* __restrict__ wmatBase,
          const float* __restrict__ dmod,
          const float* __restrict__ smul,
          const float* __restrict__ noise,
          const float* __restrict__ nsp,
          const float* __restrict__ bias,
          _Float16* __restrict__ outH,
          float* __restrict__ outF) {
  __shared__ __align__(16) _Float16 smem[2 * 6144];   // 24 KB

  const int tid  = threadIdx.x;
  const int lane = tid & 31;
  const int wv   = tid >> 5;
  const int wm   = wv >> 2;  // 0..1 : M sub-tile (32 rows)
  const int wn   = wv & 3;   // 0..3 : N sub-tile (32 cols)

  int gid = blockIdx.x;
  int n0 = (gid & 1) * 128;
  int b, t, ph, pw, y, x0, Cin, inH, inW, Ktot;
  const _Float16* wmat;
  if (mode == 0) {
    t = (gid >> 1) & 63;
    int p = (gid >> 7) & 3;
    b = gid >> 9;
    ph = p >> 1; pw = p & 1;
    Cin = 512; inH = 64; inW = 64; Ktot = 9 * 512;
    wmat = wmatBase + (size_t)p * 256 * 4608;
    y = 0; x0 = 0;
  } else {
    int mt = (gid >> 1) & 255;
    b = gid >> 9;
    int pix0 = mt * 64;
    y = pix0 >> 7; x0 = pix0 & 127;
    Cin = 256; inH = 128; inW = 128; Ktot = 9 * 256;
    wmat = wmatBase;
    t = 0; ph = 0; pw = 0;
  }
  const int HW = inH * inW;
  const _Float16* inB = inp + (size_t)b * Cin * HW;
  const int hbase = (mode == 0 ? t : y) - 1;
  const int wbase = (mode == 0 ? -1 : x0 - 1);
  const _Float16* wrow0 = wmat + (size_t)n0 * Ktot;   // weight tile row base

  v8f acc[2][2];
  const v8f vzero = {0.f, 0.f, 0.f, 0.f, 0.f, 0.f, 0.f, 0.f};
  #pragma unroll
  for (int i = 0; i < 2; ++i)
    #pragma unroll
    for (int j = 0; j < 2; ++j) acc[i][j] = vzero;

  const int am  = tid >> 2;  // A tile row (output pixel) 0..63
  const int akc = tid & 3;   // A k-chunk (8 halves)

  // gather one 8-half im2col chunk of the A tile for k-block kb
  auto gatherA = [&](int kb) -> v8h {
    int k8 = kb + akc * 8;
    int tap, i0;
    if (Cin == 512) { tap = k8 >> 9; i0 = k8 & 511; }
    else            { tap = k8 >> 8; i0 = k8 & 255; }
    int dh = tap / 3, dw = tap - dh * 3;
    int h = hbase + dh;
    int w = wbase + am + dw;
    bool ok = (h >= 0) & (h < inH) & (w >= 0) & (w < inW);
    const _Float16* src = inB + ((size_t)i0 * inH + h) * inW + w;
    v8h av;
    #pragma unroll
    for (int j = 0; j < 8; ++j) av[j] = ok ? src[(size_t)j * HW] : (_Float16)0.0f;
    return av;
  };

  // issue B-tile [128 o-rows][32 k] for k-block kb into buffer q
  auto issueB = [&](int kb, int q) {
#if defined(__has_builtin) && __has_builtin(__builtin_amdgcn_tensor_load_to_lds)
    if (tid < 32) {  // single wave drives the Tensor Data Mover
      unsigned long long ga = (unsigned long long)(uintptr_t)(wrow0 + kb);
      u32x4 g0;
      g0.x = 1u;                                     // count=1, user descriptor
      g0.y = (unsigned)(4096 + q * 12288);           // lds_addr (static LDS base = 0)
      g0.z = (unsigned)(ga & 0xFFFFFFFFull);         // global_addr[31:0]
      g0.w = (unsigned)((ga >> 32) & 0x1FFFFFFull) | (2u << 30);  // addr hi + type=2
      i32x8 g1;
      g1[0] = 1 << 16;                               // data_size = 2 bytes
      g1[1] = (int)(((unsigned)Ktot & 0xFFFFu) << 16);                  // dim0 lo16
      g1[2] = (int)((((unsigned)Ktot >> 16) & 0xFFFFu) | (256u << 16)); // dim0 hi/dim1 lo
      g1[3] = (int)(32u << 16);                      // dim1 hi=0, tile_dim0=32
      g1[4] = 128;                                   // tile_dim1=128, tile_dim2=0
      g1[5] = Ktot;                                  // tensor_dim0_stride lo32
      g1[6] = 0;
      g1[7] = 0;
      i32x4 gz4 = {0, 0, 0, 0};
      i32x8 gz8 = {0, 0, 0, 0, 0, 0, 0, 0};
      __builtin_amdgcn_tensor_load_to_lds(g0, g1, gz4, gz4, gz8, 0);
    }
#else
    #pragma unroll
    for (int r = 0; r < 2; ++r) {
      int c = tid + (r << 8);
      int nn = c >> 2, kc = c & 3;
      const _Float16* src = wrow0 + (size_t)nn * Ktot + kb + kc * 8;
      *(v8h*)(smem + q * 6144 + 2048 + nn * 32 + kc * 8) = *(const v8h*)src;
    }
#endif
  };

  const int nsteps = Ktot >> 5;

  // ---- prologue: stage k-block 0 into buffer 0
  {
    issueB(0, 0);
    v8h a0 = gatherA(0);
    *(v8h*)(smem + am * 32 + akc * 8) = a0;
#if defined(__has_builtin) && __has_builtin(__builtin_amdgcn_tensor_load_to_lds)
    if (tid < 32) __builtin_amdgcn_s_wait_tensorcnt(0);
#endif
  }
  __syncthreads();

  for (int step = 0; step < nsteps; ++step) {
    const int q = step & 1;
    const bool more = (step + 1) < nsteps;

    // ---- kick off next tile: async TDM for B, global loads for A (no wait yet)
    v8h anext;
    if (more) {
      issueB((step + 1) << 5, q ^ 1);   // DMA overlaps the WMMAs below
      anext = gatherA((step + 1) << 5); // loads in flight across compute
    }

    // ---- compute on buffer q: 2x2 tiles of v_wmma_f32_16x16x32_f16 per wave
    {
      const _Float16* Atile = smem + q * 6144;         // [64][32]
      const _Float16* Btile = smem + q * 6144 + 2048;  // [128][32] ([n][k])
      const int mrow  = lane & 15;
      const int khalf = lane >> 4;
      v16h aop[2], bop[2];
      #pragma unroll
      for (int ti = 0; ti < 2; ++ti) {
        const _Float16* pa = Atile + (wm * 32 + ti * 16 + mrow) * 32 + khalf * 8;
        v8h lo = *(const v8h*)pa;
        v8h hi = *(const v8h*)(pa + 16);
        aop[ti] = __builtin_shufflevector(lo, hi, 0,1,2,3,4,5,6,7,8,9,10,11,12,13,14,15);
      }
      #pragma unroll
      for (int tj = 0; tj < 2; ++tj) {
        const _Float16* pb = Btile + (wn * 32 + tj * 16 + mrow) * 32 + khalf * 8;
        v8h lo = *(const v8h*)pb;
        v8h hi = *(const v8h*)(pb + 16);
        bop[tj] = __builtin_shufflevector(lo, hi, 0,1,2,3,4,5,6,7,8,9,10,11,12,13,14,15);
      }
      #pragma unroll
      for (int ti = 0; ti < 2; ++ti)
        #pragma unroll
        for (int tj = 0; tj < 2; ++tj)
          acc[ti][tj] = __builtin_amdgcn_wmma_f32_16x16x32_f16(
              false, aop[ti], false, bop[tj], (short)0, acc[ti][tj], false, false);
    }

    // ---- drain next-tile staging, then swap
    if (more) {
      *(v8h*)(smem + (q ^ 1) * 6144 + am * 32 + akc * 8) = anext;
#if defined(__has_builtin) && __has_builtin(__builtin_amdgcn_tensor_load_to_lds)
      if (tid < 32) __builtin_amdgcn_s_wait_tensorcnt(0);  // after compute: overlapped
#endif
    }
    __syncthreads();
  }

  // ---- epilogue: demod, noise, bias, fused leaky relu
  const float ns = nsp[0];
  const float* noiseB = noise + (size_t)b * 128 * 128;
  #pragma unroll
  for (int ti = 0; ti < 2; ++ti)
    #pragma unroll
    for (int tj = 0; tj < 2; ++tj)
      #pragma unroll
      for (int r = 0; r < 8; ++r) {
        int mloc = wm * 32 + ti * 16 + ((lane >> 4) << 3) + r;  // 0..63
        int nn   = n0 + wn * 32 + tj * 16 + (lane & 15);        // 0..255
        int u, vv;
        if (mode == 0) { u = 2 * t + ph; vv = 2 * mloc + pw; }
        else           { u = y;          vv = x0 + mloc; }
        float val = acc[ti][tj][r] * dmod[b * 256 + nn]
                  + ns * noiseB[u * 128 + vv] + bias[nn];
        val = (val >= 0.f ? val : LEAK * val) * SQRT2;
        size_t oidx = (((size_t)b * 256 + nn) * 128 + u) * 128 + vv;
        if (mode == 0) outH[oidx] = (_Float16)(val * smul[b * 256 + nn]);
        else           outF[oidx] = val;
      }
}

// ---------------- host ----------------
extern "C" void kernel_launch(void* const* d_in, const int* in_sizes, int n_in,
                              void* d_out, int out_size, void* d_ws, size_t ws_size,
                              hipStream_t stream) {
  const float* x   = (const float*)d_in[0];
  const float* w1  = (const float*)d_in[1];
  const float* w2  = (const float*)d_in[2];
  const float* n1  = (const float*)d_in[3];
  const float* n2  = (const float*)d_in[4];
  const float* wt1 = (const float*)d_in[5];
  const float* a1w = (const float*)d_in[6];
  const float* a1b = (const float*)d_in[7];
  const float* wt2 = (const float*)d_in[8];
  const float* a2w = (const float*)d_in[9];
  const float* a2b = (const float*)d_in[10];
  const float* ns1 = (const float*)d_in[11];
  const float* ns2 = (const float*)d_in[12];
  const float* b1  = (const float*)d_in[13];
  const float* b2  = (const float*)d_in[14];

  char* ws = (char*)d_ws;
  float*    s1 = (float*)(ws + OFF_S1);
  float*    s2 = (float*)(ws + OFF_S2);
  float*    d1 = (float*)(ws + OFF_D1);
  float*    d2 = (float*)(ws + OFF_D2);
  _Float16* xs = (_Float16*)(ws + OFF_XS);
  _Float16* W1 = (_Float16*)(ws + OFF_W1);
  _Float16* W2 = (_Float16*)(ws + OFF_W2);
  _Float16* y1 = (_Float16*)(ws + OFF_Y1);
  float*   out = (float*)d_out;

  k_affine<<<24, 256, 0, stream>>>(w1, w2, a1w, a1b, a2w, a2b, s1, s2);
  k_demod<<<16, 256, 0, stream>>>(s1, s2, wt1, wt2, d1, d2);
  k_scale_x<<<65536, 256, 0, stream>>>(x, s1, xs);
  k_build_w1<<<18432, 256, 0, stream>>>(wt1, W1);
  k_build_w2<<<2304, 256, 0, stream>>>(wt2, W2);

  // conv1 (fused upsample+blur): 8b * 4 parity * 64 rows * 2 n-tiles = 4096 blocks
  conv_wmma<<<4096, 256, 0, stream>>>(0, xs, W1, d1, s2, n1, ns1, b1, y1, nullptr);
  // conv2: 8b * 256 m-tiles * 2 n-tiles = 4096 blocks
  conv_wmma<<<4096, 256, 0, stream>>>(1, y1, W2, d2, nullptr, n2, ns2, b2, nullptr, out);
}
